// FastRCNNOutputLayers_27419071218216
// MI455X (gfx1250) — compile-verified, matrix operand order checked by reading, not run on hardware
//
#include <hip/hip_runtime.h>
#include <hip/hip_bf16.h>

// ---------------- problem constants ----------------
#define R_ROWS   8192
#define D_DIM    1024
#define K_CLS    1203          // foreground classes
#define KP1      1204          // K+1 logits
#define NPAD     1216          // KP1 padded to multiple of 16 (76 tiles)
#define M_TOP    128
#define TOPK     300
#define SCORE_TH 1e-4f
#define NMS_TH   0.5f
#define NEGV     (-1e9f)
#define SCALE_CLAMP 4.135166556742356f   // log(1000/16)
#define IMG_W 1333.0f
#define IMG_H 800.0f

typedef __bf16 bf16_t;
typedef bf16_t v16bf __attribute__((ext_vector_type(16)));
typedef float  v8f   __attribute__((ext_vector_type(8)));
typedef int    v4i_t __attribute__((vector_size(16)));

union FragU { unsigned int u[8]; v16bf v; };

#if __has_builtin(__builtin_amdgcn_global_load_async_to_lds_b128)
#define HAVE_ASYNC_LDS 1
#else
#define HAVE_ASYNC_LDS 0
#endif

static __device__ __forceinline__ unsigned short f2bf(float f) {
    unsigned int u = __float_as_uint(f);
    unsigned int lsb = (u >> 16) & 1u;
    u += 0x7FFFu + lsb;                   // round-to-nearest-even
    return (unsigned short)(u >> 16);
}

// ---------------- conversion kernels ----------------
__global__ void cvt_feat_bf16(const float* __restrict__ feat,
                              unsigned short* __restrict__ featB) {
    size_t i = (size_t)blockIdx.x * blockDim.x + threadIdx.x;
    if (i < (size_t)R_ROWS * D_DIM) featB[i] = f2bf(feat[i]);
}

// wT[n][k] = bf16(w_cls[k][n]), rows n >= KP1 are zero
__global__ void cvt_wcls_T(const float* __restrict__ w_cls,
                           unsigned short* __restrict__ wT) {
    size_t i = (size_t)blockIdx.x * blockDim.x + threadIdx.x;
    if (i >= (size_t)NPAD * D_DIM) return;
    int n = (int)(i / D_DIM);
    int k = (int)(i - (size_t)n * D_DIM);
    float v = (n < KP1) ? w_cls[(size_t)k * KP1 + n] : 0.0f;
    wT[i] = f2bf(v);
}

// ---------------- WMMA GEMM: logits = featB @ wT^T ----------------
// block = 8 waves computing a 128(M) x 64(N) tile; B chunk staged in LDS
// (async-to-LDS when the toolchain exposes it), A streamed from L2/WGP$.
__global__ __launch_bounds__(256)
void gemm_wmma_bf16(const unsigned short* __restrict__ featB,
                    const unsigned short* __restrict__ wT,
                    float* __restrict__ logits) {
    const int tid  = threadIdx.x;
    const int lane = tid & 31;
    const int wv   = tid >> 5;                 // 0..7 (wave32)
    const int mBlock = blockIdx.x / 19;
    const int nGroup = blockIdx.x % 19;
    const int mBase = (mBlock * 8 + wv) << 4;  // this wave's 16 rows
    const int nBase = nGroup << 6;

    const int l15 = lane & 15;
    const int kb  = (lane >> 4) << 2;          // dword offset  (K elems 0 / 8)
    const int kb2 = (lane >> 4) << 3;          // ushort offset (K elems 0 / 8)

    // 64 rows x 128 K bf16, row pitch 136 (=128+8 pad -> bank-conflict free)
    __shared__ __align__(16) unsigned short sB[64 * 136];

    const unsigned int* arow =
        (const unsigned int*)(featB + (size_t)(mBase + l15) * D_DIM);

    v8f acc[4];
    const v8f zero = {0.f,0.f,0.f,0.f,0.f,0.f,0.f,0.f};
#pragma unroll
    for (int t = 0; t < 4; ++t) acc[t] = zero;

    for (int kc = 0; kc < 8; ++kc) {           // 8 chunks of 128 K
        __syncthreads();                       // previous chunk consumed
        // ---- stage B chunk 64x128 into LDS (1024 x 16B segments) ----
#pragma unroll
        for (int j = 0; j < 4; ++j) {
            const int seg = j * 256 + tid;     // consecutive tid -> coalesced
            const int row = seg >> 4;
            const int sIR = seg & 15;
            const unsigned short* src =
                wT + (size_t)(nBase + row) * D_DIM + kc * 128 + sIR * 8;
            unsigned short* dst = &sB[row * 136 + sIR * 8];
#if HAVE_ASYNC_LDS
            __builtin_amdgcn_global_load_async_to_lds_b128(
                (__attribute__((address_space(1))) v4i_t*)src,
                (__attribute__((address_space(3))) v4i_t*)dst, 0, 0);
#else
            uint4 v = *(const uint4*)src;
            *(uint4*)dst = v;
#endif
        }
#if HAVE_ASYNC_LDS
#if __has_builtin(__builtin_amdgcn_s_wait_asynccnt)
        __builtin_amdgcn_s_wait_asynccnt(0);
#else
        asm volatile("s_wait_asynccnt 0x0" ::: "memory");
#endif
#endif
        __syncthreads();

        // ---- compute 4 WMMA K-steps from LDS ----
#pragma unroll
        for (int k0 = 0; k0 < 128; k0 += 32) {
            const int kd = (kc * 128 + k0) >> 1;   // dword index into A row
            FragU a;
#pragma unroll
            for (int i = 0; i < 4; ++i) {
                a.u[i]     = arow[kd + kb + i];
                a.u[4 + i] = arow[kd + 8 + kb + i];
            }
#pragma unroll
            for (int t = 0; t < 4; ++t) {
                const int rb = (t * 16 + l15) * 136;
                const uint4 p0 = *(const uint4*)&sB[rb + k0 + kb2];
                const uint4 p1 = *(const uint4*)&sB[rb + k0 + 16 + kb2];
                FragU b;
                b.u[0] = p0.x; b.u[1] = p0.y; b.u[2] = p0.z; b.u[3] = p0.w;
                b.u[4] = p1.x; b.u[5] = p1.y; b.u[6] = p1.z; b.u[7] = p1.w;
                acc[t] = __builtin_amdgcn_wmma_f32_16x16x32_bf16(
                    false, a.v, false, b.v, (short)0, acc[t], false, false);
            }
        }
    }

    // C/D layout: VGPR g -> row mBase+g (+8 for lanes 16-31), col = lane&15
    const int rOff = mBase + ((lane >> 4) << 3);
#pragma unroll
    for (int t = 0; t < 4; ++t) {
        const int col = nBase + (t << 4) + l15;
#pragma unroll
        for (int g = 0; g < 8; ++g) {
            logits[(size_t)(rOff + g) * NPAD + col] = acc[t][g];
        }
    }
}

// ------- softmax + threshold, masked scores written IN PLACE (row-major) ----
__global__ __launch_bounds__(128)
void softmax_kernel(float* __restrict__ logits,
                    const float* __restrict__ b_cls) {
    const int r = blockIdx.x;
    const int tid = threadIdx.x;
    __shared__ float red[128];

    float lmax = -3.4e38f;
    for (int c = tid; c < KP1; c += 128) {
        float v = logits[(size_t)r * NPAD + c] + b_cls[c];
        lmax = fmaxf(lmax, v);
    }
    red[tid] = lmax; __syncthreads();
    for (int off = 64; off > 0; off >>= 1) {
        if (tid < off) red[tid] = fmaxf(red[tid], red[tid + off]);
        __syncthreads();
    }
    const float mx = red[0]; __syncthreads();

    float lsum = 0.f;
    for (int c = tid; c < KP1; c += 128) {
        float v = logits[(size_t)r * NPAD + c] + b_cls[c];
        lsum += __expf(v - mx);
    }
    red[tid] = lsum; __syncthreads();
    for (int off = 64; off > 0; off >>= 1) {
        if (tid < off) red[tid] += red[tid + off];
        __syncthreads();
    }
    const float inv = 1.0f / red[0];

    for (int c = tid; c < K_CLS; c += 128) {   // drop background class
        float v = logits[(size_t)r * NPAD + c] + b_cls[c];
        float s = __expf(v - mx) * inv;
        logits[(size_t)r * NPAD + c] = (s > SCORE_TH) ? s : NEGV;  // in place
    }
}

// ------- LDS-tiled 64x64 transpose: row-major scores -> class-major --------
__global__ __launch_bounds__(256)
void transpose_kernel(const float* __restrict__ scores,
                      float* __restrict__ scoresT) {
    __shared__ float tile[64][65];
    const int cb = blockIdx.x % 19;          // 19*64 = 1216 covers NPAD
    const int rb = blockIdx.x / 19;          // 128 row tiles
    const int tx  = threadIdx.x & 63;
    const int ty0 = threadIdx.x >> 6;        // 0..3

    for (int y = ty0; y < 64; y += 4) {
        tile[y][tx] = scores[(size_t)(rb * 64 + y) * NPAD + (cb * 64 + tx)];
    }
    __syncthreads();
    for (int y = ty0; y < 64; y += 4) {
        int c = cb * 64 + y;
        if (c < K_CLS)
            scoresT[(size_t)c * R_ROWS + rb * 64 + tx] = tile[tx][y];
    }
}

// ---------------- box deltas (tiny GEMM) + decode + clip ----------------
__global__ __launch_bounds__(256)
void box_kernel(const float* __restrict__ feat,
                const float* __restrict__ proposals,
                const float* __restrict__ w_box,
                const float* __restrict__ b_box,
                float* __restrict__ boxes) {
    const int lane = threadIdx.x & 31;
    const int r = blockIdx.x * 8 + (threadIdx.x >> 5);
    if (r >= R_ROWS) return;

    float d0 = 0.f, d1 = 0.f, d2 = 0.f, d3 = 0.f;
    const float* fr = feat + (size_t)r * D_DIM;
    for (int i = lane; i < D_DIM; i += 32) {
        float f = fr[i];
        d0 += f * w_box[i * 4 + 0];
        d1 += f * w_box[i * 4 + 1];
        d2 += f * w_box[i * 4 + 2];
        d3 += f * w_box[i * 4 + 3];
    }
#pragma unroll
    for (int off = 16; off > 0; off >>= 1) {
        d0 += __shfl_xor(d0, off, 32);
        d1 += __shfl_xor(d1, off, 32);
        d2 += __shfl_xor(d2, off, 32);
        d3 += __shfl_xor(d3, off, 32);
    }
    if (lane == 0) {
        d0 += b_box[0]; d1 += b_box[1]; d2 += b_box[2]; d3 += b_box[3];
        float x1 = proposals[r * 4 + 0], y1 = proposals[r * 4 + 1];
        float x2 = proposals[r * 4 + 2], y2 = proposals[r * 4 + 3];
        float w = x2 - x1, h = y2 - y1;
        float cx = x1 + 0.5f * w, cy = y1 + 0.5f * h;
        float dx = d0 * 0.1f, dy = d1 * 0.1f;
        float dw = fminf(d2 * 0.2f, SCALE_CLAMP);
        float dh = fminf(d3 * 0.2f, SCALE_CLAMP);
        float pcx = dx * w + cx, pcy = dy * h + cy;
        float pw = __expf(dw) * w, ph = __expf(dh) * h;
        float ox1 = pcx - 0.5f * pw, oy1 = pcy - 0.5f * ph;
        float ox2 = pcx + 0.5f * pw, oy2 = pcy + 0.5f * ph;
        boxes[r * 4 + 0] = fminf(fmaxf(ox1, 0.f), IMG_W);
        boxes[r * 4 + 1] = fminf(fmaxf(oy1, 0.f), IMG_H);
        boxes[r * 4 + 2] = fminf(fmaxf(ox2, 0.f), IMG_W);
        boxes[r * 4 + 3] = fminf(fmaxf(oy2, 0.f), IMG_H);
    }
}

// ---------------- per-class top-128 + NMS ----------------
__global__ __launch_bounds__(128)
void nms_kernel(const float* __restrict__ scoresT,
                const float* __restrict__ boxes,
                float* __restrict__ per_scores,
                int* __restrict__ per_idx) {
    const int c = blockIdx.x;
    const int tid = threadIdx.x;

    __shared__ float sc[R_ROWS];        // 32 KB class column in LDS
    __shared__ float rv[128];
    __shared__ int   ri[128];
    __shared__ float selV[M_TOP];
    __shared__ int   selI[M_TOP];
    __shared__ float bx[M_TOP][4];
    __shared__ int   keep[M_TOP];
    __shared__ int   supFlag;

    const float* col = scoresT + (size_t)c * R_ROWS;
    for (int j = tid; j < R_ROWS; j += 128) sc[j] = col[j];
    __syncthreads();

    // iterative argmax selection of top-128
    for (int s = 0; s < M_TOP; ++s) {
        float bv = -3.4e38f; int bi = 0;
        for (int j = tid; j < R_ROWS; j += 128) {
            float v = sc[j];
            if (v > bv) { bv = v; bi = j; }
        }
        rv[tid] = bv; ri[tid] = bi; __syncthreads();
        for (int off = 64; off > 0; off >>= 1) {
            if (tid < off) {
                float vo = rv[tid + off];
                if (vo > rv[tid] || (vo == rv[tid] && ri[tid + off] < ri[tid])) {
                    rv[tid] = vo; ri[tid] = ri[tid + off];
                }
            }
            __syncthreads();
        }
        if (tid == 0) {
            selV[s] = rv[0]; selI[s] = ri[0];
            sc[ri[0]] = -3.4e38f;          // remove from candidates
        }
        __syncthreads();
    }

    // gather boxes of selected proposals
    {
        int idx = selI[tid];
        bx[tid][0] = boxes[idx * 4 + 0];
        bx[tid][1] = boxes[idx * 4 + 1];
        bx[tid][2] = boxes[idx * 4 + 2];
        bx[tid][3] = boxes[idx * 4 + 3];
        keep[tid] = 0;
    }
    __syncthreads();

    const float ax1 = bx[tid][0], ay1 = bx[tid][1], ax2 = bx[tid][2], ay2 = bx[tid][3];
    const float areaA = (ax2 - ax1) * (ay2 - ay1);

    for (int i = 0; i < M_TOP; ++i) {
        if (tid == 0) supFlag = 0;
        __syncthreads();
        if (keep[tid]) {
            float bx1 = bx[i][0], by1 = bx[i][1], bx2 = bx[i][2], by2 = bx[i][3];
            float ix = fmaxf(0.f, fminf(ax2, bx2) - fmaxf(ax1, bx1));
            float iy = fmaxf(0.f, fminf(ay2, by2) - fmaxf(ay1, by1));
            float inter = ix * iy;
            float areaB = (bx2 - bx1) * (by2 - by1);
            float iou = inter / (areaA + areaB - inter + 1e-6f);
            if (iou > NMS_TH) atomicOr(&supFlag, 1);
        }
        __syncthreads();
        if (tid == i) {
            keep[i] = (selV[i] > NEGV * 0.5f) && (supFlag == 0);
        }
        __syncthreads();
    }

    per_scores[(size_t)c * M_TOP + tid] = keep[tid] ? selV[tid] : NEGV;
    per_idx[(size_t)c * M_TOP + tid] = selI[tid];
}

// ---------------- global top-300 + output assembly ----------------
__global__ __launch_bounds__(1024)
void topk_kernel(const float* __restrict__ per_scores,
                 const int* __restrict__ per_idx,
                 const float* __restrict__ boxes,
                 float* __restrict__ work,
                 float* __restrict__ out) {
    const int tid = threadIdx.x;
    const int N = K_CLS * M_TOP;
    __shared__ float rv[1024];
    __shared__ int   ri[1024];

    for (int i = tid; i < N; i += 1024) work[i] = per_scores[i];
    __threadfence();
    __syncthreads();

    for (int t = 0; t < TOPK; ++t) {
        float bv = -3.4e38f; int bi = 0;
        for (int i = tid; i < N; i += 1024) {
            float v = work[i];
            if (v > bv) { bv = v; bi = i; }
        }
        rv[tid] = bv; ri[tid] = bi; __syncthreads();
        for (int off = 512; off > 0; off >>= 1) {
            if (tid < off) {
                float vo = rv[tid + off];
                if (vo > rv[tid] || (vo == rv[tid] && ri[tid + off] < ri[tid])) {
                    rv[tid] = vo; ri[tid] = ri[tid + off];
                }
            }
            __syncthreads();
        }
        if (tid == 0) {
            int fi = ri[0];
            float val = rv[0];
            int cls = fi / M_TOP;
            int prop = per_idx[fi];
            out[t * 6 + 0] = boxes[prop * 4 + 0];
            out[t * 6 + 1] = boxes[prop * 4 + 1];
            out[t * 6 + 2] = boxes[prop * 4 + 2];
            out[t * 6 + 3] = boxes[prop * 4 + 3];
            out[t * 6 + 4] = val;
            out[t * 6 + 5] = (float)cls;
            work[fi] = -3.4e38f;
        }
        __threadfence();
        __syncthreads();
    }
}

// ---------------- launcher ----------------
extern "C" void kernel_launch(void* const* d_in, const int* in_sizes, int n_in,
                              void* d_out, int out_size, void* d_ws, size_t ws_size,
                              hipStream_t stream) {
    const float* feat      = (const float*)d_in[0];
    const float* proposals = (const float*)d_in[1];
    const float* w_cls     = (const float*)d_in[2];
    const float* b_cls     = (const float*)d_in[3];
    const float* w_box     = (const float*)d_in[4];
    const float* b_box     = (const float*)d_in[5];
    float* out = (float*)d_out;

    char* ws = (char*)d_ws;
    constexpr size_t FEATB_OFF  = 0;
    constexpr size_t FEATB_SZ   = (size_t)R_ROWS * D_DIM * 2;
    constexpr size_t WT_OFF     = FEATB_OFF + FEATB_SZ;
    constexpr size_t WT_SZ      = (size_t)NPAD * D_DIM * 2;
    constexpr size_t LOG_OFF    = WT_OFF + WT_SZ;
    constexpr size_t LOG_SZ     = (size_t)R_ROWS * NPAD * 4;
    constexpr size_t SCT_OFF    = LOG_OFF + LOG_SZ;
    constexpr size_t SCT_SZ     = (size_t)K_CLS * R_ROWS * 4;
    constexpr size_t BOX_OFF    = SCT_OFF + SCT_SZ;
    constexpr size_t BOX_SZ     = (size_t)R_ROWS * 4 * 4;
    constexpr size_t PSC_OFF    = BOX_OFF + BOX_SZ;
    constexpr size_t PSC_SZ     = (size_t)K_CLS * M_TOP * 4;
    constexpr size_t PIX_OFF    = PSC_OFF + PSC_SZ;
    constexpr size_t PIX_SZ     = (size_t)K_CLS * M_TOP * 4;
    constexpr size_t WORK_OFF   = PIX_OFF + PIX_SZ;

    unsigned short* featB = (unsigned short*)(ws + FEATB_OFF);
    unsigned short* wT    = (unsigned short*)(ws + WT_OFF);
    float* logits         = (float*)(ws + LOG_OFF);   // becomes masked scores
    float* scoresT        = (float*)(ws + SCT_OFF);
    float* boxes          = (float*)(ws + BOX_OFF);
    float* per_scores     = (float*)(ws + PSC_OFF);
    int*   per_idx        = (int*)(ws + PIX_OFF);
    float* work           = (float*)(ws + WORK_OFF);

    // 1) bf16 conversions
    {
        size_t n = (size_t)R_ROWS * D_DIM;
        cvt_feat_bf16<<<(unsigned)((n + 255) / 256), 256, 0, stream>>>(feat, featB);
    }
    {
        size_t n = (size_t)NPAD * D_DIM;
        cvt_wcls_T<<<(unsigned)((n + 255) / 256), 256, 0, stream>>>(w_cls, wT);
    }
    // 2) WMMA GEMM: 64 M-blocks * 19 N-groups, 128x64 per block, B via LDS
    gemm_wmma_bf16<<<64 * 19, 256, 0, stream>>>(featB, wT, logits);
    // 3) softmax + threshold, masked scores in place (row-major)
    softmax_kernel<<<R_ROWS, 128, 0, stream>>>(logits, b_cls);
    // 4) coalesced transpose to class-major
    transpose_kernel<<<19 * 128, 256, 0, stream>>>(logits, scoresT);
    // 5) box decode
    box_kernel<<<R_ROWS / 8, 256, 0, stream>>>(feat, proposals, w_box, b_box, boxes);
    // 6) per-class top-128 + NMS
    nms_kernel<<<K_CLS, 128, 0, stream>>>(scoresT, boxes, per_scores, per_idx);
    // 7) global top-300 + assemble (300 x 6)
    topk_kernel<<<1, 1024, 0, stream>>>(per_scores, per_idx, boxes, work, out);
}
